// SAINTLite_42125039239564
// MI455X (gfx1250) — matrix operand order repaired and checked by reading, not compiled, not run on hardware
//
#include <hip/hip_runtime.h>
#include <math.h>

typedef __attribute__((ext_vector_type(16))) _Float16 v16h;
typedef __attribute__((ext_vector_type(8)))  _Float16 h8;
typedef __attribute__((ext_vector_type(8)))  float    v8f;

#define BB   4
#define SS   2048
#define FIN  8
#define DD   256
#define HH   8
#define LL   2
#define FFN  1024
#define DHH  32
#define NSKK 2000
#define BSR  (BB*SS)   // 8192 rows

// ---------------------------------------------------------------------------
// Fragment loaders (layouts per CDNA5 ISA 7.12.2, wave32)
// A 16x32 f16 frag: lane row M=lane&15, kg=lane>>4:
//   elems 0..7 = K kg*8+0..7, elems 8..15 = K kg*8+16..23
// B 32x16 f16 frag: lane col N=lane&15, kg=lane>>4: same K order as A (strided gather).
// ---------------------------------------------------------------------------
static __device__ __forceinline__ v16h load_a_frag(const _Float16* __restrict__ ap) {
  h8 lo = *(const h8*)(ap + 0);    // K kg*8 + 0..7
  h8 hi = *(const h8*)(ap + 16);   // K kg*8 + 16..23
  return __builtin_shufflevector(lo, hi, 0,1,2,3,4,5,6,7,8,9,10,11,12,13,14,15);
}

static __device__ __forceinline__ v16h load_b_frag(const _Float16* __restrict__ bp, int ld) {
  v16h r;
#pragma unroll
  for (int e = 0; e < 16; e++) r[e] = bp[(size_t)(e < 8 ? e : e + 8) * ld];
  return r;
}

// ---------------------------------------------------------------------------
// GEMM: C[M,N] = act(A[M,K] @ B[K,N] + bias). One wave per block, 32 rows x
// NT*16 cols per wave: two A fragments share every B fragment (8 WMMAs per
// K-step at NT=4). f16 in, f32 accumulate. TC: float | _Float16 (C dtype).
// act: 0 = none, 1 = exact GELU. Requires M%32==0, K%32==0.
// ---------------------------------------------------------------------------
template <typename TC, int NT>
__global__ void __launch_bounds__(32) wmma_gemm(
    const _Float16* __restrict__ A, const _Float16* __restrict__ B,
    const float* __restrict__ bias, TC* __restrict__ C,
    int K, int lda, int ldb, int ldc, int act, int nbase)
{
  const int lane = threadIdx.x & 31;
  const int hg = lane >> 4;
  const int lr = lane & 15;
  const int m0 = blockIdx.x * 32;
  const int n0 = nbase + blockIdx.y * (NT * 16);

  v8f p0_0 = {}, p0_1 = {}, p0_2 = {}, p0_3 = {};   // rows m0..m0+15
  v8f p1_0 = {}, p1_1 = {}, p1_2 = {}, p1_3 = {};   // rows m0+16..m0+31
  const _Float16* arow0 = A + (size_t)(m0 + lr) * lda + hg * 8;
  const _Float16* arow1 = arow0 + (size_t)16 * lda;

  for (int k0 = 0; k0 < K; k0 += 32) {
    const v16h afA = load_a_frag(arow0 + k0);
    const v16h afB = load_a_frag(arow1 + k0);
    const _Float16* bp = B + (size_t)(k0 + hg * 8) * ldb + n0 + lr;
    if (k0 + 32 < K) {     // prefetch next K-step (global_prefetch_b8)
      __builtin_prefetch(arow0 + k0 + 32, 0, 3);
      __builtin_prefetch(bp + (size_t)32 * ldb, 0, 3);
    }
    {
      const v16h bf = load_b_frag(bp, ldb);
      p0_0 = __builtin_amdgcn_wmma_f32_16x16x32_f16(false, afA, false, bf, (short)0, p0_0, false, false);
      p1_0 = __builtin_amdgcn_wmma_f32_16x16x32_f16(false, afB, false, bf, (short)0, p1_0, false, false);
    }
    if (NT > 1) {
      const v16h bf = load_b_frag(bp + 16, ldb);
      p0_1 = __builtin_amdgcn_wmma_f32_16x16x32_f16(false, afA, false, bf, (short)0, p0_1, false, false);
      p1_1 = __builtin_amdgcn_wmma_f32_16x16x32_f16(false, afB, false, bf, (short)0, p1_1, false, false);
    }
    if (NT > 2) {
      const v16h bf = load_b_frag(bp + 32, ldb);
      p0_2 = __builtin_amdgcn_wmma_f32_16x16x32_f16(false, afA, false, bf, (short)0, p0_2, false, false);
      p1_2 = __builtin_amdgcn_wmma_f32_16x16x32_f16(false, afB, false, bf, (short)0, p1_2, false, false);
    }
    if (NT > 3) {
      const v16h bf = load_b_frag(bp + 48, ldb);
      p0_3 = __builtin_amdgcn_wmma_f32_16x16x32_f16(false, afA, false, bf, (short)0, p0_3, false, false);
      p1_3 = __builtin_amdgcn_wmma_f32_16x16x32_f16(false, afB, false, bf, (short)0, p1_3, false, false);
    }
  }

  auto store_tile = [&](const v8f& av, int t, int mbase) {
    const int n = n0 + t * 16 + lr;
    const float bvv = bias ? bias[n] : 0.0f;
#pragma unroll
    for (int j = 0; j < 8; j++) {
      const int m = mbase + hg * 8 + j;
      float vv = av[j] + bvv;
      if (act == 1) vv = 0.5f * vv * (1.0f + erff(vv * 0.70710678118654752f));
      C[(size_t)m * ldc + n] = (TC)vv;
    }
  };
  store_tile(p0_0, 0, m0); store_tile(p1_0, 0, m0 + 16);
  if (NT > 1) { store_tile(p0_1, 1, m0); store_tile(p1_1, 1, m0 + 16); }
  if (NT > 2) { store_tile(p0_2, 2, m0); store_tile(p1_2, 2, m0 + 16); }
  if (NT > 3) { store_tile(p0_3, 3, m0); store_tile(p1_3, 3, m0 + 16); }
}

// ---------------------------------------------------------------------------
// KERPLE bias table: tab[(l*H+h)*S + d] = -softplus(p) * log1p(softplus(a)*d)
// ---------------------------------------------------------------------------
__global__ void build_bias_kernel(const float* __restrict__ kp,
                                  const float* __restrict__ ka,
                                  float* __restrict__ tab)
{
  const int lh = blockIdx.y;                        // 0 .. L*H-1
  const int d  = blockIdx.x * 256 + threadIdx.x;    // 0 .. S-1
  const float pp = log1pf(expf(kp[lh]));
  const float aa = log1pf(expf(ka[lh]));
  tab[(size_t)lh * SS + d] = -pp * log1pf(aa * (float)d);
}

// ---------------------------------------------------------------------------
// Flash attention, one wave per (b, h, 16-row query block).
// Computes S^T = K_block x Q^T so the per-chunk operand (K) uses contiguous
// 16B loads; the strided Q^T fragment is loaded once. Per-lane scalar softmax
// stats (lane owns query column s = q0 + (lane&15)). KERPLE bias from table.
// q/k/v/o layout: (B, S, H, DH) f16. f32 softmax.
// ---------------------------------------------------------------------------
__global__ void __launch_bounds__(32) attn_kernel(
    const _Float16* __restrict__ q, const _Float16* __restrict__ k,
    const _Float16* __restrict__ v, _Float16* __restrict__ o,
    const float* __restrict__ btab, int layer)
{
  __shared__ _Float16 pl[16 * 32];   // P~ staging (S^T C-layout -> P A-layout)

  const int lane = threadIdx.x & 31;
  const int hg = lane >> 4;
  const int lr = lane & 15;
  const int q0 = blockIdx.x * 16;
  const int h  = blockIdx.y;
  const int b  = blockIdx.z;

  const float invs = 0.17677669529663687f;             // 1/sqrt(32)
  const float* bt = btab + (size_t)(layer * HH + h) * SS;

  // Q^T B-fragment: lane = query column, elems = dh in A's K order (loaded once)
  const _Float16* qtp = q + (size_t)(b * SS + q0 + lr) * 256 + h * DHH + hg * 8;
  v16h qt;
#pragma unroll
  for (int e = 0; e < 16; e++) qt[e] = qtp[e < 8 ? e : e + 8];

  v8f o0 = {}, o1 = {};
  float mrow = -3.0e38f, lsum = 0.0f;
  const int s_col = q0 + lr;          // query this lane owns for stats

  const int tmax = q0 + 15;
  for (int t0 = 0; t0 <= tmax; t0 += 32) {
    // K A-fragments: rows = keys (contiguous 16B loads)
    const _Float16* ka0 = k + (size_t)(b * SS + t0 + lr) * 256 + h * DHH + hg * 8;
    const v16h kf0 = load_a_frag(ka0);
    const v16h kf1 = load_a_frag(ka0 + (size_t)16 * 256);
    const v8f z = {};
    // S^T tiles: rows t = t0 + hg*8 + j (+16), cols s = q0 + lr
    v8f s0v = __builtin_amdgcn_wmma_f32_16x16x32_f16(false, kf0, false, qt, (short)0, z, false, false);
    v8f s1v = __builtin_amdgcn_wmma_f32_16x16x32_f16(false, kf1, false, qt, (short)0, z, false, false);

    // scale + KERPLE bias (table) + causal mask; chunk max over t
    float sc0[8], sc1[8];
    float cm = -3.0e38f;
#pragma unroll
    for (int j = 0; j < 8; j++) {
      const int d0 = s_col - (t0 + hg * 8 + j);
      const int d1 = d0 - 16;
      float b0 = bt[d0 < 0 ? 0 : d0];
      float b1 = bt[d1 < 0 ? 0 : d1];
      if (d0 < 0) b0 = -1.0e9f;
      if (d1 < 0) b1 = -1.0e9f;
      sc0[j] = s0v[j] * invs + b0;
      sc1[j] = s1v[j] * invs + b1;
      cm = fmaxf(cm, fmaxf(sc0[j], sc1[j]));
    }
    cm = fmaxf(cm, __shfl_xor(cm, 16, 32));

    // online softmax update (per-lane scalars)
    const float mn  = fmaxf(mrow, cm);
    const float scl = __expf(mrow - mn);
    mrow = mn;
    float rs = 0.0f;
#pragma unroll
    for (int j = 0; j < 8; j++) {
      sc0[j] = __expf(sc0[j] - mn);
      sc1[j] = __expf(sc1[j] - mn);
      rs += sc0[j] + sc1[j];
    }
    rs += __shfl_xor(rs, 16, 32);
    lsum = lsum * scl + rs;

    // P~ to LDS in A-layout: row s=lr, K=t_rel; two contiguous 16B runs
    h8 pk0, pk1;
#pragma unroll
    for (int j = 0; j < 8; j++) { pk0[j] = (_Float16)sc0[j]; pk1[j] = (_Float16)sc1[j]; }
    __syncthreads();
    *(h8*)&pl[lr * 32 + hg * 8]      = pk0;
    *(h8*)&pl[lr * 32 + 16 + hg * 8] = pk1;
    __syncthreads();
    const v16h pa = load_a_frag((const _Float16*)&pl[lr * 32 + hg * 8]);

    // rescale O accumulators (per-row factor broadcast from stats lanes)
#pragma unroll
    for (int j = 0; j < 8; j++) {
      const float sj = __shfl(scl, hg * 8 + j, 32);
      o0[j] *= sj;
      o1[j] *= sj;
    }

    // V B-fragments: lane = dh column, elems = key rows in A's K order (strided)
    const _Float16* vp = v + (size_t)(b * SS + t0 + hg * 8) * 256 + h * DHH + lr;
    o0 = __builtin_amdgcn_wmma_f32_16x16x32_f16(false, pa, false, load_b_frag(vp, 256),      (short)0, o0, false, false);
    o1 = __builtin_amdgcn_wmma_f32_16x16x32_f16(false, pa, false, load_b_frag(vp + 16, 256), (short)0, o1, false, false);
  }

#pragma unroll
  for (int j = 0; j < 8; j++) {
    const float ls = __shfl(lsum, hg * 8 + j, 32);
    const float inv = 1.0f / ls;
    const int s = q0 + hg * 8 + j;
    _Float16* op = o + (size_t)(b * SS + s) * 256 + h * DHH;
    op[lr]      = (_Float16)(o0[j] * inv);
    op[16 + lr] = (_Float16)(o1[j] * inv);
  }
}

// ---------------------------------------------------------------------------
// LN(x + res): writes f32 stream (next residual) + f16 mirror (WMMA A input)
// one wave per row of 256
// ---------------------------------------------------------------------------
__global__ void __launch_bounds__(32) add_ln_kernel(
    const float* __restrict__ x, const float* __restrict__ r,
    const float* __restrict__ g, const float* __restrict__ bt,
    float* __restrict__ out32, _Float16* __restrict__ out16)
{
  const int row = blockIdx.x;
  const int lane = threadIdx.x & 31;
  const float* xp = x + (size_t)row * 256 + lane * 8;
  const float* rp = r + (size_t)row * 256 + lane * 8;
  float4 x0 = ((const float4*)xp)[0], x1 = ((const float4*)xp)[1];
  float4 r0 = ((const float4*)rp)[0], r1 = ((const float4*)rp)[1];
  float vals[8] = { x0.x + r0.x, x0.y + r0.y, x0.z + r0.z, x0.w + r0.w,
                    x1.x + r1.x, x1.y + r1.y, x1.z + r1.z, x1.w + r1.w };
  float s = 0.0f;
#pragma unroll
  for (int j = 0; j < 8; j++) s += vals[j];
#pragma unroll
  for (int d = 1; d < 32; d <<= 1) s += __shfl_xor(s, d, 32);
  const float mean = s * (1.0f / 256.0f);
  float vs = 0.0f;
#pragma unroll
  for (int j = 0; j < 8; j++) { const float dd = vals[j] - mean; vs += dd * dd; }
#pragma unroll
  for (int d = 1; d < 32; d <<= 1) vs += __shfl_xor(vs, d, 32);
  const float rstd = rsqrtf(vs * (1.0f / 256.0f) + 1e-6f);
  h8 hv;
#pragma unroll
  for (int j = 0; j < 8; j++) {
    const int c = lane * 8 + j;
    const float vv = g[c] * (vals[j] - mean) * rstd + bt[c];
    out32[(size_t)row * 256 + c] = vv;
    hv[j] = (_Float16)vv;
  }
  *(h8*)&out16[(size_t)row * 256 + lane * 8] = hv;
}

// ---------------------------------------------------------------------------
// x += tanh(elapsed * time_W + time_b) + (s/S) * pos_W + pos_b ; mirror to f16
// ---------------------------------------------------------------------------
__global__ void add_enc_kernel(
    float* __restrict__ x, _Float16* __restrict__ x16,
    const float* __restrict__ ff,
    const float* __restrict__ tW, const float* __restrict__ tb,
    const float* __restrict__ pW, const float* __restrict__ pb)
{
  const int s = blockIdx.x, b = blockIdx.y, d = threadIdx.x;
  const float el = ff[(size_t)(b * SS + s) * FIN];
  const float te = tanhf(el * tW[d] + tb[d]);
  const float pe = ((float)s / (float)SS) * pW[d] + pb[d];
  const size_t idx = (size_t)(b * SS + s) * 256 + d;
  const float vv = x[idx] + te + pe;
  x[idx] = vv;
  x16[idx] = (_Float16)vv;
}

// ---------------------------------------------------------------------------
// Build concat([inter_emb[idx], pattern_emb[idx], ff @ feat_W + feat_b]) rows
// as f16 (WMMA A operand of the input projection).
// ---------------------------------------------------------------------------
__global__ void build_xin_kernel(
    const int* __restrict__ inter, const int* __restrict__ pat,
    const float* __restrict__ ff, const float* __restrict__ ie,
    const float* __restrict__ pe, const float* __restrict__ fW,
    const float* __restrict__ fb, _Float16* __restrict__ xin)
{
  const int r = blockIdx.x;
  const int c = threadIdx.x;   // 0..383
  float vv;
  if (c < 256) {
    vv = ie[(size_t)inter[r] * 256 + c];
  } else if (c < 320) {
    vv = pe[(size_t)pat[r] * 64 + (c - 256)];
  } else {
    const int cc = c - 320;
    vv = fb[cc];
#pragma unroll
    for (int f = 0; f < 8; f++) vv += ff[(size_t)r * 8 + f] * fW[f * 64 + cc];
  }
  xin[(size_t)r * 384 + c] = (_Float16)vv;
}

// ---------------------------------------------------------------------------
// f32 -> f16 weight conversion (one element per thread)
// ---------------------------------------------------------------------------
__global__ void cvt_kernel(const float* __restrict__ src,
                           _Float16* __restrict__ dst, int n)
{
  int i = blockIdx.x * 256 + threadIdx.x;
  if (i < n) dst[i] = (_Float16)src[i];
}

// ---------------------------------------------------------------------------
extern "C" void kernel_launch(void* const* d_in, const int* in_sizes, int n_in,
                              void* d_out, int out_size, void* d_ws, size_t ws_size,
                              hipStream_t stream)
{
  (void)in_sizes; (void)n_in; (void)out_size; (void)ws_size;
  const int*   interactions = (const int*)  d_in[0];
  const int*   patterns     = (const int*)  d_in[1];
  const float* ff           = (const float*)d_in[2];
  const float* inter_emb    = (const float*)d_in[3];
  const float* pattern_emb  = (const float*)d_in[4];
  const float* feat_W       = (const float*)d_in[5];
  const float* feat_b       = (const float*)d_in[6];
  const float* in_W         = (const float*)d_in[7];
  const float* in_b         = (const float*)d_in[8];
  const float* time_W       = (const float*)d_in[9];
  const float* time_b       = (const float*)d_in[10];
  const float* pos_W        = (const float*)d_in[11];
  const float* pos_b        = (const float*)d_in[12];
  const float* Wq           = (const float*)d_in[13];
  const float* bq           = (const float*)d_in[14];
  const float* Wk           = (const float*)d_in[15];
  const float* bk           = (const float*)d_in[16];
  const float* Wv           = (const float*)d_in[17];
  const float* bv           = (const float*)d_in[18];
  const float* Wo           = (const float*)d_in[19];
  const float* bo           = (const float*)d_in[20];
  const float* kerple_p     = (const float*)d_in[21];
  const float* kerple_a     = (const float*)d_in[22];
  const float* ln1_g        = (const float*)d_in[23];
  const float* ln1_b        = (const float*)d_in[24];
  const float* ln2_g        = (const float*)d_in[25];
  const float* ln2_b        = (const float*)d_in[26];
  const float* ffn_W1       = (const float*)d_in[27];
  const float* ffn_b1       = (const float*)d_in[28];
  const float* ffn_W2       = (const float*)d_in[29];
  const float* ffn_b2       = (const float*)d_in[30];
  const float* out_W        = (const float*)d_in[31];
  const float* out_b        = (const float*)d_in[32];

  // ---- workspace layout ----
  char* p = (char*)d_ws;
  float* x    = (float*)p;            p += (size_t)BSR * 256 * 4;
  float* tmp  = (float*)p;            p += (size_t)BSR * 256 * 4;
  float* btab = (float*)p;            p += (size_t)LL * HH * SS * 4;
  _Float16* x16   = (_Float16*)p;     p += (size_t)BSR * 256 * 2;
  _Float16* q16   = (_Float16*)p;     p += (size_t)BSR * 256 * 2;
  _Float16* k16   = (_Float16*)p;     p += (size_t)BSR * 256 * 2;
  _Float16* v16_  = (_Float16*)p;     p += (size_t)BSR * 256 * 2;
  _Float16* o16   = (_Float16*)p;     p += (size_t)BSR * 256 * 2;
  _Float16* xin16 = (_Float16*)p;     p += (size_t)BSR * 384 * 2;
  _Float16* h16   = (_Float16*)p;     p += (size_t)BSR * 1024 * 2;
  _Float16* w_in  = (_Float16*)p;     p += (size_t)384 * 256 * 2;
  _Float16* w_q   = (_Float16*)p;     p += (size_t)LL * 256 * 256 * 2;
  _Float16* w_k   = (_Float16*)p;     p += (size_t)LL * 256 * 256 * 2;
  _Float16* w_v   = (_Float16*)p;     p += (size_t)LL * 256 * 256 * 2;
  _Float16* w_o   = (_Float16*)p;     p += (size_t)LL * 256 * 256 * 2;
  _Float16* w_f1  = (_Float16*)p;     p += (size_t)LL * 256 * 1024 * 2;
  _Float16* w_f2  = (_Float16*)p;     p += (size_t)LL * 1024 * 256 * 2;
  _Float16* w_out = (_Float16*)p;     p += (size_t)256 * 2000 * 2;

  const dim3 wv(32);
  const int MT = BSR / 32;   // 256 row blocks (32 rows per wave)

  // ---- prepass: weight conversion + KERPLE bias table ----
  {
    int n;
    n = 384 * 256;          cvt_kernel<<<(n + 255) / 256, 256, 0, stream>>>(in_W,   w_in,  n);
    n = LL * 256 * 256;     cvt_kernel<<<(n + 255) / 256, 256, 0, stream>>>(Wq,     w_q,   n);
    n = LL * 256 * 256;     cvt_kernel<<<(n + 255) / 256, 256, 0, stream>>>(Wk,     w_k,   n);
    n = LL * 256 * 256;     cvt_kernel<<<(n + 255) / 256, 256, 0, stream>>>(Wv,     w_v,   n);
    n = LL * 256 * 256;     cvt_kernel<<<(n + 255) / 256, 256, 0, stream>>>(Wo,     w_o,   n);
    n = LL * 256 * 1024;    cvt_kernel<<<(n + 255) / 256, 256, 0, stream>>>(ffn_W1, w_f1,  n);
    n = LL * 1024 * 256;    cvt_kernel<<<(n + 255) / 256, 256, 0, stream>>>(ffn_W2, w_f2,  n);
    n = 256 * 2000;         cvt_kernel<<<(n + 255) / 256, 256, 0, stream>>>(out_W,  w_out, n);
    build_bias_kernel<<<dim3(SS / 256, LL * HH), 256, 0, stream>>>(kerple_p, kerple_a, btab);
  }

  // 1. build concatenated input rows (f16)
  build_xin_kernel<<<dim3(BSR), dim3(384), 0, stream>>>(
      interactions, patterns, ff, inter_emb, pattern_emb, feat_W, feat_b, xin16);

  // 2. x = Xin @ in_W + in_b  (f32 C; f16 mirror made by add_enc)
  wmma_gemm<float, 4><<<dim3(MT, 4), wv, 0, stream>>>(
      xin16, w_in, in_b, x, 384, 384, 256, 256, 0, 0);

  // 3. time / positional encodings (writes x f32 + x16 mirror)
  add_enc_kernel<<<dim3(SS, BB), dim3(256), 0, stream>>>(
      x, x16, ff, time_W, time_b, pos_W, pos_b);

  for (int i = 0; i < LL; i++) {
    const size_t wo256 = (size_t)i * 256 * 256;

    // QKV from f16 mirror
    wmma_gemm<_Float16, 4><<<dim3(MT, 4), wv, 0, stream>>>(
        x16, w_q + wo256, bq + i * 256, q16, 256, 256, 256, 256, 0, 0);
    wmma_gemm<_Float16, 4><<<dim3(MT, 4), wv, 0, stream>>>(
        x16, w_k + wo256, bk + i * 256, k16, 256, 256, 256, 256, 0, 0);
    wmma_gemm<_Float16, 4><<<dim3(MT, 4), wv, 0, stream>>>(
        x16, w_v + wo256, bv + i * 256, v16_, 256, 256, 256, 256, 0, 0);

    attn_kernel<<<dim3(SS / 16, HH, BB), wv, 0, stream>>>(
        q16, k16, v16_, o16, btab, i);

    // O projection -> tmp (f32)
    wmma_gemm<float, 4><<<dim3(MT, 4), wv, 0, stream>>>(
        o16, w_o + wo256, bo + i * 256, tmp, 256, 256, 256, 256, 0, 0);
    add_ln_kernel<<<dim3(BSR), wv, 0, stream>>>(
        x, tmp, ln1_g + i * 256, ln1_b + i * 256, x, x16);

    // FFN1 (+GELU) from f16 mirror -> h16
    wmma_gemm<_Float16, 4><<<dim3(MT, 16), wv, 0, stream>>>(
        x16, w_f1 + (size_t)i * 256 * 1024, ffn_b1 + i * 1024, h16,
        256, 256, 1024, 1024, 1, 0);
    // FFN2 -> tmp (f32)
    wmma_gemm<float, 4><<<dim3(MT, 4), wv, 0, stream>>>(
        h16, w_f2 + (size_t)i * 1024 * 256, ffn_b2 + i * 256, tmp,
        1024, 1024, 256, 256, 0, 0);
    add_ln_kernel<<<dim3(BSR), wv, 0, stream>>>(
        x, tmp, ln2_g + i * 256, ln2_b + i * 256, x, x16);
  }

  // 4. output head: (8192 x 256) @ (256 x 2000) from f16 mirror
  //    31 full 64-col blocks + one 16-col tail at n=1984
  wmma_gemm<float, 4><<<dim3(MT, 31), wv, 0, stream>>>(
      x16, w_out, out_b, (float*)d_out, 256, 256, 2000, 2000, 0, 0);
  wmma_gemm<float, 1><<<dim3(MT, 1), wv, 0, stream>>>(
      x16, w_out, out_b, (float*)d_out, 256, 256, 2000, 2000, 0, 1984);
}
